// Gloria_7215545057306
// MI455X (gfx1250) — matrix operand order repaired and checked by reading, not compiled, not run on hardware
//
#include <hip/hip_runtime.h>

typedef __attribute__((ext_vector_type(16))) __bf16        bf16x16;
typedef __attribute__((ext_vector_type(8)))  float         f32x8;
typedef __attribute__((ext_vector_type(8)))  unsigned int  u32x8;

#define B_DIM 256
#define D_DIM 768
#define Q_DIM 96
#define S_DIM 361
#define SPAD  384          // S padded to multiple of 32
#define KC    32           // WMMA K per bf16 op
#define KST   40           // ushort stride of transposed ctx stage rows (80B: 16B-aligned, bank-skewed)
#define NKC1  (D_DIM / KC) // 24 k-chunks over ndf (GEMM1)
#define NKC2  (SPAD / KC)  // 12 k-chunks over S   (GEMM2)
#define NT    (Q_DIM / 16) // 6 q tiles
#define MT2   (D_DIM / 16) // 48 d tiles
#define NWAVE 8
#define TPB   256
#define TEMP1 4.0f

__device__ __forceinline__ unsigned short f2bfbits(float f) {
  unsigned u = __builtin_bit_cast(unsigned, f);
  u += 0x7FFFu + ((u >> 16) & 1u);           // round-to-nearest-even
  return (unsigned short)(u >> 16);
}
__device__ __forceinline__ __bf16 bfbits(unsigned short h) {
  return __builtin_bit_cast(__bf16, h);
}

// A-matrix 16x32 bf16 fragment from LDS stage laid out [s][k] (row stride KST ushorts).
// ISA layout: lane L: m = L%16, half h = L/16; k pairs are contiguous -> two 16B LDS loads.
__device__ __forceinline__ bf16x16 load_a_lds_t(const unsigned short* src, int mbase, int lane) {
  int m = mbase + (lane & 15);
  int h = (lane >> 4) << 3;                 // 0 or 8
  const unsigned short* row = src + m * KST + h;
  uint4 lo = *(const uint4*)(row);          // k = h..h+7
  uint4 hi = *(const uint4*)(row + 16);     // k = 16+h..16+h+7
  u32x8 t;
  t[0] = lo.x; t[1] = lo.y; t[2] = lo.z; t[3] = lo.w;
  t[4] = hi.x; t[5] = hi.y; t[6] = hi.z; t[7] = hi.w;
  return __builtin_bit_cast(bf16x16, t);
}

// B-matrix 32x16 bf16 fragment from LDS laid out [k][n] (stride Q_DIM ushorts).
// Layout: lane L -> N = nbase+L%16 ; element s holds K = kbase+8*(L/16)+s (s<8), kbase+16+8*(L/16)+s-8 (s>=8).
__device__ __forceinline__ bf16x16 load_b_lds(const unsigned short* bsrc, int kbase, int nbase, int lane) {
  int n = nbase + (lane & 15);
  int h = (lane >> 4) << 3;
  bf16x16 t;
#pragma unroll
  for (int s = 0; s < 16; ++s) {
    int k = (s < 8) ? (h + s) : (16 + h + (s - 8));
    t[s] = bfbits(bsrc[(kbase + k) * Q_DIM + n]);
  }
  return t;
}

// A-matrix 16x32 bf16 fragment straight from global f32 ctx row (m = d fixed per lane, L2-resident).
__device__ __forceinline__ bf16x16 load_a_gbl(const float* __restrict__ ctxrow, int s0, int lane) {
  int h = (lane >> 4) << 3;
  bf16x16 a;
#pragma unroll
  for (int v = 0; v < 8; ++v) {
    int k = (v < 4) ? (h + 2 * v) : (16 + h + 2 * (v - 4));
    int s = s0 + k;
    float f0 = (s     < S_DIM) ? ctxrow[s]     : 0.f;
    float f1 = (s + 1 < S_DIM) ? ctxrow[s + 1] : 0.f;
    a[2 * v]     = bfbits(f2bfbits(f0));
    a[2 * v + 1] = bfbits(f2bfbits(f1));
  }
  return a;
}

__global__ __launch_bounds__(TPB) void gloria_attn_kernel(
    const float* __restrict__ qry, const float* __restrict__ ctx, float* __restrict__ out) {
  extern __shared__ unsigned char smem[];
  unsigned short* ctx_s   = (unsigned short*)smem;                     // [SPAD][KST] bf16 (transposed)
  unsigned short* q_s     = ctx_s + SPAD * KST;                        // [KC][Q_DIM] bf16
  float*          attn_s  = (float*)(q_s + KC * Q_DIM);                // [SPAD][Q_DIM] f32
  unsigned short* attnb_s = (unsigned short*)(attn_s + SPAD * Q_DIM);  // [SPAD][Q_DIM] bf16

  const int tid  = threadIdx.x;
  const int lane = tid & 31;
  const int wave = tid >> 5;
  const int b    = blockIdx.x;

  const float* ctx_b = ctx + (size_t)b * D_DIM * S_DIM;
  const float* qry_b = qry + (size_t)b * D_DIM * Q_DIM;

  // ---------------- Phase 1: attn[s,q] = sum_d ctx[d,s] * query[d,q] ----------------
  f32x8 zero = {};
  f32x8 acc[3 * NT];
#pragma unroll
  for (int i = 0; i < 3 * NT; ++i) acc[i] = zero;

  for (int kc = 0; kc < NKC1; ++kc) {
    // stage ctx chunk transposed: ctx_s[s][r] = bf16(ctx[kc*32+r][s]); zero pad s>=361
    for (int idx = tid; idx < KC * SPAD; idx += TPB) {
      int r = idx / SPAD, s = idx - r * SPAD;
      float v = (s < S_DIM) ? ctx_b[(size_t)(kc * KC + r) * S_DIM + s] : 0.f;
      ctx_s[s * KST + r] = f2bfbits(v);
    }
    // stage query chunk [32][96] with vectorized f32x4 reads, packed 8B stores
    for (int idx = tid; idx < KC * (Q_DIM / 4); idx += TPB) {
      int r = idx / (Q_DIM / 4), q4 = idx - r * (Q_DIM / 4);
      int q = 4 * q4;
      float4 v = *(const float4*)(qry_b + (size_t)(kc * KC + r) * Q_DIM + q);
      unsigned lo = (unsigned)f2bfbits(v.x) | ((unsigned)f2bfbits(v.y) << 16);
      unsigned hi = (unsigned)f2bfbits(v.z) | ((unsigned)f2bfbits(v.w) << 16);
      uint2 pk; pk.x = lo; pk.y = hi;
      *(uint2*)(q_s + r * Q_DIM + q) = pk;
    }
    // prefetch next chunk of ctx into cache (global_prefetch_b8)
    if (kc + 1 < NKC1) {
      int r = tid >> 3;
      int soff = (tid & 7) * 45;
      __builtin_prefetch(ctx_b + (size_t)((kc + 1) * KC + r) * S_DIM + soff, 0, 1);
    }
    __syncthreads();

    bf16x16 afrag[3];
#pragma unroll
    for (int mi = 0; mi < 3; ++mi)
      afrag[mi] = load_a_lds_t(ctx_s, (wave + NWAVE * mi) * 16, lane);
#pragma unroll
    for (int qt = 0; qt < NT; ++qt) {
      bf16x16 bfrag = load_b_lds(q_s, 0, qt * 16, lane);
#pragma unroll
      for (int mi = 0; mi < 3; ++mi) {
        acc[mi * NT + qt] = __builtin_amdgcn_wmma_f32_16x16x32_bf16(
            false, afrag[mi], false, bfrag, (short)0, acc[mi * NT + qt], false, false);
      }
    }
    __syncthreads();
  }

  // spill accumulator tiles into LDS attn buffer (C layout: n=lane%16, m=r+8*(lane/16))
#pragma unroll
  for (int mi = 0; mi < 3; ++mi) {
    int mt = wave + NWAVE * mi;
#pragma unroll
    for (int qt = 0; qt < NT; ++qt) {
#pragma unroll
      for (int r = 0; r < 8; ++r) {
        int srow = mt * 16 + r + ((lane >> 4) << 3);
        int qcol = qt * 16 + (lane & 15);
        attn_s[srow * Q_DIM + qcol] = acc[mi * NT + qt][r];
      }
    }
  }
  __syncthreads();

  // ---------------- Phase 2: softmax over q for each s row ----------------
  for (int s = tid; s < S_DIM; s += TPB) {
    float* row = attn_s + s * Q_DIM;
    float m = -3.0e38f;
    for (int q = 0; q < Q_DIM; ++q) m = fmaxf(m, row[q]);
    float sum = 0.f;
    for (int q = 0; q < Q_DIM; ++q) { float e = __expf(row[q] - m); row[q] = e; sum += e; }
    float inv = 1.f / sum;
    for (int q = 0; q < Q_DIM; ++q) row[q] *= inv;
  }
  __syncthreads();

  // ---------------- Phase 3: x4 temperature, softmax over s per q; emit attn_map ----------------
  if (tid < Q_DIM) {
    int q = tid;
    float m = -3.0e38f;
    for (int s = 0; s < S_DIM; ++s) m = fmaxf(m, attn_s[s * Q_DIM + q]);
    m *= TEMP1;
    float sum = 0.f;
    for (int s = 0; s < S_DIM; ++s) {
      float e = __expf(TEMP1 * attn_s[s * Q_DIM + q] - m);
      attn_s[s * Q_DIM + q] = e;        // column owned by this thread: safe in-place
      sum += e;
    }
    float inv = 1.f / sum;
    float* aout = out + (size_t)B_DIM * D_DIM * Q_DIM + ((size_t)b * Q_DIM + q) * S_DIM;
    for (int s = 0; s < S_DIM; ++s) {
      float p = attn_s[s * Q_DIM + q] * inv;
      aout[s] = p;
      attnb_s[s * Q_DIM + q] = f2bfbits(p);
    }
    for (int s = S_DIM; s < SPAD; ++s) attnb_s[s * Q_DIM + q] = 0;  // padded K must be zero
  }
  __syncthreads();

  // ---------------- Phase 4: wc[d,q] = sum_s ctx[d,s] * attn2[q,s] ----------------
  for (int mi = 0; mi < MT2 / NWAVE; ++mi) {   // 6 d-tiles per wave
    int mt = wave + NWAVE * mi;
    const float* ctxrow = ctx_b + (size_t)(mt * 16 + (lane & 15)) * S_DIM;
    f32x8 acc2[NT];
#pragma unroll
    for (int qt = 0; qt < NT; ++qt) acc2[qt] = zero;
    for (int kc = 0; kc < NKC2; ++kc) {
      bf16x16 a = load_a_gbl(ctxrow, kc * KC, lane);
#pragma unroll
      for (int qt = 0; qt < NT; ++qt) {
        bf16x16 bfr = load_b_lds(attnb_s, kc * KC, qt * 16, lane);
        acc2[qt] = __builtin_amdgcn_wmma_f32_16x16x32_bf16(
            false, a, false, bfr, (short)0, acc2[qt], false, false);
      }
    }
#pragma unroll
    for (int qt = 0; qt < NT; ++qt) {
#pragma unroll
      for (int r = 0; r < 8; ++r) {
        int d = mt * 16 + r + ((lane >> 4) << 3);
        int q = qt * 16 + (lane & 15);
        out[((size_t)b * D_DIM + d) * Q_DIM + q] = acc2[qt][r];
      }
    }
  }
}

extern "C" void kernel_launch(void* const* d_in, const int* in_sizes, int n_in,
                              void* d_out, int out_size, void* d_ws, size_t ws_size,
                              hipStream_t stream) {
  (void)in_sizes; (void)n_in; (void)d_ws; (void)ws_size; (void)out_size;
  const float* qry = (const float*)d_in[0];   // [256,768,96]  f32
  const float* ctx = (const float*)d_in[1];   // [256,768,19,19] f32
  float* out = (float*)d_out;                 // wc [256,768,96] ++ attn [256,96,361]

  size_t shmem = (size_t)SPAD * KST * sizeof(unsigned short)          // ctx stage (transposed)
               + (size_t)KC * Q_DIM * sizeof(unsigned short)          // query stage
               + (size_t)SPAD * Q_DIM * sizeof(float)                 // attn f32
               + (size_t)SPAD * Q_DIM * sizeof(unsigned short);       // attn bf16

  gloria_attn_kernel<<<dim3(B_DIM), dim3(TPB), shmem, stream>>>(qry, ctx, out);
}
